// MPNNGNN_29326036697881
// MI455X (gfx1250) — matrix operand order repaired and checked by reading, not compiled, not run on hardware
//
#include <hip/hip_runtime.h>
#include <hip/hip_bf16.h>

#define V_N 12500
#define E_N 200000
#define H_N 32
#define NSTEPS 6

typedef __bf16 bf16;
typedef __attribute__((ext_vector_type(16))) __bf16 v16bf;
typedef __attribute__((ext_vector_type(8)))  float  v8f;

// ---------------------------------------------------------------------------
// Projection: h = relu(nf @ w1 + b1) @ w2 + b2   (V,32)->(V,32)
// ---------------------------------------------------------------------------
__global__ void proj_kernel(const float* __restrict__ nf,
                            const float* __restrict__ w1, const float* __restrict__ b1,
                            const float* __restrict__ w2, const float* __restrict__ b2,
                            float* __restrict__ h) {
    __shared__ float sx[8 * 32];
    __shared__ float st[8 * 32];
    int t = threadIdx.x;
    int nl = t >> 5, c = t & 31;
    int node = blockIdx.x * 8 + nl;
    float x = (node < V_N) ? nf[node * 32 + c] : 0.f;
    sx[nl * 32 + c] = x;
    __syncthreads();
    float acc = b1[c];
#pragma unroll
    for (int k = 0; k < 32; ++k) acc += sx[nl * 32 + k] * w1[k * 32 + c];
    st[nl * 32 + c] = fmaxf(acc, 0.f);
    __syncthreads();
    float acc2 = b2[c];
#pragma unroll
    for (int i = 0; i < 32; ++i) acc2 += st[nl * 32 + i] * w2[i * 32 + c];
    if (node < V_N) h[node * 32 + c] = acc2;
}

// ---------------------------------------------------------------------------
// Edge hidden: Z[e][k] = relu(edge_feats[e] . e_w1[:,k] + e_b1[k]), stored
// bf16 in WMMA 16-bit A-matrix swizzle (A tile = 16 edges x 32 k per chunk):
//   K(elem,lane) = (elem&7) + ((elem>>3)<<4) + ((lane>>4)<<3), M = lane&15
// ---------------------------------------------------------------------------
__global__ void edge_hidden_kernel(const float* __restrict__ ef,
                                   const float* __restrict__ w1,
                                   const float* __restrict__ b1,
                                   bf16* __restrict__ zswz) {
    int e = blockIdx.x;       // edge
    int k = threadIdx.x;      // 0..127
    float acc = b1[k];
#pragma unroll
    for (int i = 0; i < 16; ++i) acc += ef[e * 16 + i] * w1[i * 128 + k];
    float z = fmaxf(acc, 0.f);
    int T = e >> 4, c = k >> 5, kk = k & 31;
    int lane_hi = (kk >> 3) & 1;
    int elem = (kk & 7) + ((kk >> 4) << 3);
    int lane = (lane_hi << 4) | (e & 15);
    zswz[(((size_t)T * 4 + c) * 32 + lane) * 16 + elem] = (bf16)z;
}

// ---------------------------------------------------------------------------
// e_w2 (128 x 1024 f32) -> bf16, WMMA 16-bit B-matrix swizzle per (k-chunk,
// n-tile): lanes 0-15 K=0..7,16..23 (N=lane), lanes 16-31 K=8..15,24..31.
// ---------------------------------------------------------------------------
__global__ void w2_swizzle_kernel(const float* __restrict__ w2, bf16* __restrict__ bwz) {
    int id = blockIdx.x * blockDim.x + threadIdx.x;
    if (id >= 128 * 1024) return;
    int k = id >> 10, n = id & 1023;
    int c = k >> 5, kk = k & 31;
    int lane = (((kk >> 3) & 1) << 4) | (n & 15);
    int elem = (kk & 7) + ((kk >> 4) << 3);
    int nt = n >> 4;
    bwz[((size_t)(c * 64 + nt) * 32 + lane) * 16 + elem] = (bf16)w2[id];
}

__global__ void zero_kernel(float* __restrict__ p, int n) {
    int i = blockIdx.x * blockDim.x + threadIdx.x;
    if (i < n) p[i] = 0.f;
}

// ---------------------------------------------------------------------------
// Fused NNConv: per 16-edge tile, M = Z_tile(16x128) @ e_w2(128x1024) via
// v_wmma_f32_16x16x32_bf16; each 16x16 output tile covers a single input
// channel i = col>>5, so the epilogue scales by h[src][i] and accumulates
// msg in VGPRs. 16 waves: wave w owns o-half (w&1), i in {w>>1,+8,+16,+24}.
// B operands (e_w2 tiles) are T-loop invariant -> held in 128 VGPRs/wave.
// Cross-wave reduction in LDS (deterministic), f32 global atomic scatter.
// ---------------------------------------------------------------------------
__global__ __launch_bounds__(512, 1) void nnconv_kernel(
        const bf16* __restrict__ zswz, const bf16* __restrict__ bwz,
        const float* __restrict__ eb2, const float* __restrict__ h,
        const int* __restrict__ src, const int* __restrict__ dst,
        float* __restrict__ agg) {
    __shared__ float hs[16 * 32];        //  2 KB: gathered h[src] rows
    __shared__ float msgp[16 * 16 * 16]; // 16 KB: per-wave partial messages

    const int t    = threadIdx.x;
    const int lane = t & 31;
    const int w    = t >> 5;
    const int oh    = w & 1;    // which 16-column half of the 32 outputs
    const int ibase = w >> 1;   // base input channel (stride 8 over 4 iters)

    // Hoist loop-invariant B operands (this wave's 16 e_w2 tiles) and bias.
    v16bf b[4][4];
    float eb[4];
#pragma unroll
    for (int it = 0; it < 4; ++it) {
        const int i_idx = ibase + it * 8;
        const int nt    = i_idx * 2 + oh;
        eb[it] = eb2[i_idx * 32 + oh * 16 + (lane & 15)];
#pragma unroll
        for (int c = 0; c < 4; ++c)
            b[it][c] = *(const v16bf*)(bwz + ((size_t)(c * 64 + nt) * 32 + lane) * 16);
    }

    for (int T = blockIdx.x; T < E_N / 16; T += gridDim.x) {
        {   // gather h[src] rows for this tile (512 threads -> 512 values)
            int e = t >> 5, i = t & 31;
            hs[e * 32 + i] = h[(size_t)src[T * 16 + e] * 32 + i];
        }
        __syncthreads();

        // A operands (Z tile), shared by this wave's 4 output tiles.
        v16bf a0 = *(const v16bf*)(zswz + (((size_t)T * 4 + 0) * 32 + lane) * 16);
        v16bf a1 = *(const v16bf*)(zswz + (((size_t)T * 4 + 1) * 32 + lane) * 16);
        v16bf a2 = *(const v16bf*)(zswz + (((size_t)T * 4 + 2) * 32 + lane) * 16);
        v16bf a3 = *(const v16bf*)(zswz + (((size_t)T * 4 + 3) * 32 + lane) * 16);

        float macc[8] = {0.f, 0.f, 0.f, 0.f, 0.f, 0.f, 0.f, 0.f};

#pragma unroll
        for (int it = 0; it < 4; ++it) {
            const int i_idx = ibase + it * 8;   // input channel of this tile
            v8f acc = {};
            acc = __builtin_amdgcn_wmma_f32_16x16x32_bf16(false, a0, false, b[it][0], (short)0, acc, false, false);
            acc = __builtin_amdgcn_wmma_f32_16x16x32_bf16(false, a1, false, b[it][1], (short)0, acc, false, false);
            acc = __builtin_amdgcn_wmma_f32_16x16x32_bf16(false, a2, false, b[it][2], (short)0, acc, false, false);
            acc = __builtin_amdgcn_wmma_f32_16x16x32_bf16(false, a3, false, b[it][3], (short)0, acc, false, false);
#pragma unroll
            for (int j = 0; j < 8; ++j) {
                int edge = j + ((lane >> 4) << 3);   // C layout: M = j (+8 upper half)
                macc[j] += (acc[j] + eb[it]) * hs[edge * 32 + i_idx];
            }
        }

        // write per-wave partial msg to LDS
#pragma unroll
        for (int j = 0; j < 8; ++j) {
            int edge = j + ((lane >> 4) << 3);
            msgp[(w * 16 + edge) * 16 + (lane & 15)] = macc[j];
        }
        __syncthreads();

        // deterministic reduce over the 8 waves sharing each o-half + scatter
        {
            int e = t >> 5, o = t & 31, ohh = o >> 4, ol = o & 15;
            float s = 0.f;
#pragma unroll
            for (int r = 0; r < 8; ++r)
                s += msgp[((ohh + r * 2) * 16 + e) * 16 + ol];
            float* dstp = &agg[(size_t)dst[T * 16 + e] * 32 + o];
            __hip_atomic_fetch_add(dstp, s, __ATOMIC_RELAXED, __HIP_MEMORY_SCOPE_AGENT);
        }
        __syncthreads();
    }
}

// ---------------------------------------------------------------------------
// GRU step: x = relu(agg + conv_bias); gates from x and hidden; update hcur.
// ---------------------------------------------------------------------------
__global__ void gru_kernel(const float* __restrict__ agg, const float* __restrict__ cbias,
                           const float* __restrict__ wih, const float* __restrict__ whh,
                           const float* __restrict__ bih, const float* __restrict__ bhh,
                           float* __restrict__ hcur) {
    __shared__ float swih[96 * 32];
    __shared__ float swhh[96 * 32];
    __shared__ float sx[8 * 32];
    __shared__ float sh[8 * 32];
    int t = threadIdx.x;
    for (int i = t; i < 96 * 32; i += 256) { swih[i] = wih[i]; swhh[i] = whh[i]; }
    int nl = t >> 5, o = t & 31;
    int node = blockIdx.x * 8 + nl;
    float x = 0.f, hp = 0.f;
    if (node < V_N) {
        x  = fmaxf(agg[node * 32 + o] + cbias[o], 0.f);
        hp = hcur[node * 32 + o];
    }
    sx[nl * 32 + o] = x;
    sh[nl * 32 + o] = hp;
    __syncthreads();
    float gir = bih[o], giz = bih[o + 32], gin = bih[o + 64];
    float ghr = bhh[o], ghz = bhh[o + 32], ghn = bhh[o + 64];
#pragma unroll
    for (int i = 0; i < 32; ++i) {
        float xi = sx[nl * 32 + i], hi = sh[nl * 32 + i];
        gir += xi * swih[(o)      * 32 + i];
        giz += xi * swih[(o + 32) * 32 + i];
        gin += xi * swih[(o + 64) * 32 + i];
        ghr += hi * swhh[(o)      * 32 + i];
        ghz += hi * swhh[(o + 32) * 32 + i];
        ghn += hi * swhh[(o + 64) * 32 + i];
    }
    float r  = 1.f / (1.f + __expf(-(gir + ghr)));
    float z  = 1.f / (1.f + __expf(-(giz + ghz)));
    float ng = tanhf(gin + r * ghn);
    if (node < V_N) hcur[node * 32 + o] = (1.f - z) * ng + z * hp;
}

// ---------------------------------------------------------------------------
// Decoder: out = relu(h @ w1 + b1) @ w2 + b2   (V,32)->(V,64)
// ---------------------------------------------------------------------------
__global__ void dec_kernel(const float* __restrict__ h,
                           const float* __restrict__ w1, const float* __restrict__ b1,
                           const float* __restrict__ w2, const float* __restrict__ b2,
                           float* __restrict__ out) {
    __shared__ float st[4 * 32];
    int t = threadIdx.x;
    if (t < 128) {
        int nl = t >> 5, i = t & 31;
        int node = blockIdx.x * 4 + nl;
        float acc = b1[i];
        if (node < V_N) {
#pragma unroll
            for (int k = 0; k < 32; ++k) acc += h[node * 32 + k] * w1[k * 32 + i];
        }
        st[nl * 32 + i] = fmaxf(acc, 0.f);
    }
    __syncthreads();
    int nl = t >> 6, o = t & 63;
    int node = blockIdx.x * 4 + nl;
    if (node < V_N) {
        float acc = b2[o];
#pragma unroll
        for (int i = 0; i < 32; ++i) acc += st[nl * 32 + i] * w2[i * 64 + o];
        out[node * 64 + o] = acc;
    }
}

// ---------------------------------------------------------------------------
extern "C" void kernel_launch(void* const* d_in, const int* in_sizes, int n_in,
                              void* d_out, int out_size, void* d_ws, size_t ws_size,
                              hipStream_t stream) {
    const float* node_feats = (const float*)d_in[0];
    const float* edge_feats = (const float*)d_in[1];
    const int*   src        = (const int*)  d_in[2];
    const int*   dst        = (const int*)  d_in[3];
    const float* proj_w1 = (const float*)d_in[4];
    const float* proj_b1 = (const float*)d_in[5];
    const float* proj_w2 = (const float*)d_in[6];
    const float* proj_b2 = (const float*)d_in[7];
    const float* e_w1    = (const float*)d_in[8];
    const float* e_b1    = (const float*)d_in[9];
    const float* e_w2    = (const float*)d_in[10];
    const float* e_b2    = (const float*)d_in[11];
    const float* cbias   = (const float*)d_in[12];
    const float* gru_wih = (const float*)d_in[13];
    const float* gru_whh = (const float*)d_in[14];
    const float* gru_bih = (const float*)d_in[15];
    const float* gru_bhh = (const float*)d_in[16];
    const float* dec_w1  = (const float*)d_in[17];
    const float* dec_b1  = (const float*)d_in[18];
    const float* dec_w2  = (const float*)d_in[19];
    const float* dec_b2  = (const float*)d_in[20];
    float* out = (float*)d_out;

    // workspace layout
    char* ws = (char*)d_ws;
    float* hcur = (float*)ws;                               // V*32 f32 = 1.6 MB
    float* agg  = (float*)(ws + (size_t)V_N * 32 * 4);      // V*32 f32 = 1.6 MB
    bf16*  zswz = (bf16*)(ws + (size_t)V_N * 32 * 4 * 2);   // E*128 bf16 = 51.2 MB
    bf16*  bwz  = (bf16*)(ws + (size_t)V_N * 32 * 4 * 2 + (size_t)E_N * 128 * 2); // 256 KB

    (void)in_sizes; (void)n_in; (void)out_size; (void)ws_size;

    // one-time precompute
    w2_swizzle_kernel<<<(128 * 1024 + 255) / 256, 256, 0, stream>>>(e_w2, bwz);
    proj_kernel<<<(V_N + 7) / 8, 256, 0, stream>>>(node_feats, proj_w1, proj_b1,
                                                   proj_w2, proj_b2, hcur);
    edge_hidden_kernel<<<E_N, 128, 0, stream>>>(edge_feats, e_w1, e_b1, zswz);

    // message-passing steps
    for (int s = 0; s < NSTEPS; ++s) {
        zero_kernel<<<(V_N * 32 + 255) / 256, 256, 0, stream>>>(agg, V_N * 32);
        nnconv_kernel<<<1250, 512, 0, stream>>>(zswz, bwz, e_b2, hcur,
                                                src, dst, agg);
        gru_kernel<<<(V_N + 7) / 8, 256, 0, stream>>>(agg, cbias, gru_wih, gru_whh,
                                                      gru_bih, gru_bhh, hcur);
    }

    dec_kernel<<<(V_N + 3) / 4, 256, 0, stream>>>(hcur, dec_w1, dec_b1,
                                                  dec_w2, dec_b2, out);
}
